// ConsecutiveMatch_53231824666982
// MI455X (gfx1250) — matrix operand order repaired and checked by reading, not compiled, not run on hardware
//
#include <hip/hip_runtime.h>
#include <hip/hip_bf16.h>
#include <stdint.h>

// ConsecutiveMatch on MI455X (gfx1250, wave32, WMMA + TDM):
//   pcos[t] = normalize(desc[t]) @ normalize(desc[t+1])^T   (7 x 2048x2048x256)
//   confidence = rowmax, matched = points[t+1][rowargmax]
// Compute-bound vs HBM, but naive tiling is L2-bound on B re-reads; this
// version shares each B tile across 8 waves via LDS, staged by the Tensor
// Data Mover (double-buffered), and splits the column range 4-way for
// occupancy with a tiny final reduce.

typedef __attribute__((ext_vector_type(16))) _Float16     v16h;
typedef __attribute__((ext_vector_type(8)))  float        v8f;
typedef __attribute__((ext_vector_type(4)))  unsigned int v4u;
typedef __attribute__((ext_vector_type(8)))  int          v8i_t;
typedef __attribute__((ext_vector_type(4)))  int          v4i_t;

#define T_FRAMES 8
#define N_PTS    2048
#define D_DIM    256
#define K_STEPS  (D_DIM / 32)        // 8 wmma k-steps per 16x16 tile
#define N_JT     (N_PTS / 16)        // 128 column tiles
#define N_IT     (N_PTS / 16)        // 128 row tiles
#define NWAVES   8                   // waves per block; each owns one i-tile
#define JCHUNKS  4                   // column-range split for occupancy
#define JT_PER_CHUNK (N_JT / JCHUNKS)   // 32
#define IB_PER_T (N_IT / NWAVES)        // 16 row-blocks per frame pair
#define LDS_ROW_H 264                // padded LDS row stride in halves (528B)

#if __has_builtin(__builtin_amdgcn_tensor_load_to_lds)
#define USE_TDM 1
#else
#define USE_TDM 0
#endif

union HFrag {
    v16h     v;
    _Float16 h[16];
    uint4    q[2];   // two 16-byte chunks (8 halves each)
};

// ---------------------------------------------------------------------------
// Kernel 1: row-normalize to f16.  One wave32 per row of 256 floats.
// scale = 1 / max(||x||, 1e-8)  (torch CosineSimilarity eps semantics)
// ---------------------------------------------------------------------------
__global__ __launch_bounds__(256) void normalize_f16_kernel(
    const float* __restrict__ desc, _Float16* __restrict__ out)
{
    const int wid  = threadIdx.x >> 5;
    const int lane = threadIdx.x & 31;
    const size_t row = (size_t)blockIdx.x * NWAVES + wid;   // 0 .. T*N-1

    const float* src = desc + row * D_DIM + lane * 8;
    float4 a = *(const float4*)(src);
    float4 b = *(const float4*)(src + 4);
    float ss = a.x * a.x + a.y * a.y + a.z * a.z + a.w * a.w
             + b.x * b.x + b.y * b.y + b.z * b.z + b.w * b.w;
    for (int m = 16; m >= 1; m >>= 1) ss += __shfl_xor(ss, m, 32);
    const float scale = 1.0f / fmaxf(sqrtf(ss), 1e-8f);

    union { _Float16 h[8]; uint4 q; } o;
    o.h[0] = (_Float16)(a.x * scale);
    o.h[1] = (_Float16)(a.y * scale);
    o.h[2] = (_Float16)(a.z * scale);
    o.h[3] = (_Float16)(a.w * scale);
    o.h[4] = (_Float16)(b.x * scale);
    o.h[5] = (_Float16)(b.y * scale);
    o.h[6] = (_Float16)(b.z * scale);
    o.h[7] = (_Float16)(b.w * scale);
    *(uint4*)(out + row * D_DIM + lane * 8) = o.q;
}

// ---------------------------------------------------------------------------
// TDM: stage one 16-row x 256-half B tile into LDS with padded rows.
// D# per CDNA5 ISA §8.3-8.5:
//   group0: count=1 | lds_addr | global_addr(57b) | type=2
//   group1: data_size=2B, pad_enable, pad_interval=128 DWORDs (code 6),
//           pad_amount=4 DWORDs (code 3) -> 512B row + 16B pad = 528B stride,
//           tensor 256 x 2048 (dim0_stride 256), tile 256 x 16.
// This toolchain (clang-23/therock) exposes the 6-arg builtin form:
//   (v4u g0, v8i g1, v4i g2, v4i g3, v8i g4, i32 cpol)
// Groups 2/3/4 are don't-care for a 2D tile (tile_dim2 == 0) -> zero-filled.
// ---------------------------------------------------------------------------
#if USE_TDM
__device__ __forceinline__ void tdm_stage_tile(const _Float16* gtile,
                                               _Float16* ldst)
{
    const uint64_t ga  = (uint64_t)(uintptr_t)gtile;
    const uint32_t lda = (uint32_t)(uintptr_t)ldst;  // low 32 bits = LDS offset

    v4u g0;
    g0.x = 1u;                                        // count=1, user mode
    g0.y = lda;                                       // lds_addr
    g0.z = (uint32_t)ga;                              // global_addr[31:0]
    g0.w = (uint32_t)(ga >> 32) | 0x80000000u;        // addr[56:32] | type=2

    v8i_t g1;
    g1[0] = (int)((1u << 16)      // data_size = 2 bytes
                | (1u << 20)      // pad_enable
                | (6u << 22)      // pad_interval: 128 DWORDs (512B)
                | (3u << 25));    // pad_amount:   4 DWORDs (16B)
    g1[1] = (int)(256u << 16);    // tensor_dim0[15:0] = 256
    g1[2] = (int)(2048u << 16);   // tensor_dim0[31:16]=0 | tensor_dim1[15:0]=2048
    g1[3] = (int)(256u << 16);    // tensor_dim1[31:16]=0 | tile_dim0 = 256
    g1[4] = 16;                   // tile_dim1 = 16, tile_dim2 = 0
    g1[5] = 256;                  // tensor_dim0_stride = 256 elements
    g1[6] = 0;
    g1[7] = 0;

    v4i_t z4 = {0, 0, 0, 0};                 // groups 2/3 unused (2D tile)
    v8i_t z8 = {0, 0, 0, 0, 0, 0, 0, 0};     // extra group in 6-arg form
    __builtin_amdgcn_tensor_load_to_lds(g0, g1, z4, z4, z8, 0);
}
#endif

// ---------------------------------------------------------------------------
// Kernel 2: WMMA similarity + fused row max/argmax over one column chunk.
// Block = 8 waves; wave w owns i-tile (ib*8+w) and keeps its A fragments in
// registers for the whole chunk.  All waves walk the same j tile, reading B
// fragments from an LDS tile staged by the TDM (double-buffered).
//
// Fragment layouts per CDNA5 ISA 7.12.2 (wave32, 16-bit):
//   A 16x32: lane m(<16) row m holds K {k0..k0+7, k0+16..k0+23}; lane m+16
//            holds K {k0+8..k0+15, k0+24..k0+31}.
//   B 32x16: lane n(<16) = column n (dst row n), K {k0..k0+7, k0+16..k0+23};
//            lane n+16 same column, K {k0+8..k0+15, k0+24..k0+31}.
//   C/D f32: slot v @ lane -> row = v + 8*(lane>=16), col = lane&15.
// ---------------------------------------------------------------------------
__global__ __launch_bounds__(256) void match_wmma_kernel(
    const _Float16* __restrict__ nd,    // normalized f16 [T][N][D]
    float* __restrict__ pval,           // [T-1][JCHUNKS][N] partial max
    int*   __restrict__ pidx)           // [T-1][JCHUNKS][N] partial argmax
{
    __shared__ _Float16 sB[2][16 * LDS_ROW_H];

    const int wid  = threadIdx.x >> 5;
    const int lane = threadIdx.x & 31;
    const int rm   = lane & 15;
    const int hsel = lane >> 4;

    const int b  = blockIdx.x;
    const int t  = b / (IB_PER_T * JCHUNKS);
    const int r  = b % (IB_PER_T * JCHUNKS);
    const int ib = r / JCHUNKS;
    const int jc = r % JCHUNKS;

    const _Float16* srcM = nd + (size_t)t       * N_PTS * D_DIM;
    const _Float16* dstM = nd + (size_t)(t + 1) * N_PTS * D_DIM;

    const int itile = ib * NWAVES + wid;

    // --- A fragments: 16 rows x K=256, resident in VGPRs for whole chunk ---
    HFrag afrag[K_STEPS];
    {
        const _Float16* ar = srcM + (size_t)(itile * 16 + rm) * D_DIM;
        #pragma unroll
        for (int ks = 0; ks < K_STEPS; ++ks) {
            const int k0 = ks * 32 + hsel * 8;
            afrag[ks].q[0] = *(const uint4*)(ar + k0);
            afrag[ks].q[1] = *(const uint4*)(ar + k0 + 16);
        }
    }

    float best[8];
    int   bestj[8];
    #pragma unroll
    for (int v = 0; v < 8; ++v) { best[v] = -3.0e38f; bestj[v] = 0; }

    const int jt0 = jc * JT_PER_CHUNK;

#if USE_TDM
    if (wid == 0) {
        tdm_stage_tile(dstM + (size_t)(jt0 * 16) * D_DIM, &sB[0][0]);
        __builtin_amdgcn_s_wait_tensorcnt(0);
    }
    __syncthreads();
#else
    for (int c = threadIdx.x; c < 512; c += 256) {
        const int row = c >> 5, off = (c & 31) * 8;
        *(uint4*)(&sB[0][row * LDS_ROW_H + off]) =
            *(const uint4*)(dstM + (size_t)(jt0 * 16 + row) * D_DIM + off);
    }
    __syncthreads();
#endif

    for (int s = 0; s < JT_PER_CHUNK; ++s) {
        const int jt  = jt0 + s;
        const int buf = s & 1;

#if USE_TDM
        if (wid == 0 && s + 1 < JT_PER_CHUNK) {   // prefetch next tile
            tdm_stage_tile(dstM + (size_t)((jt + 1) * 16) * D_DIM,
                           &sB[buf ^ 1][0]);
        }
#endif
        // --- preload all B fragments from LDS, then the WMMA chain ---
        const _Float16* bl = &sB[buf][rm * LDS_ROW_H + hsel * 8];
        HFrag bfrag[K_STEPS];
        #pragma unroll
        for (int ks = 0; ks < K_STEPS; ++ks) {
            bfrag[ks].q[0] = *(const uint4*)(bl + ks * 32);
            bfrag[ks].q[1] = *(const uint4*)(bl + ks * 32 + 16);
        }
        v8f acc = {};
        #pragma unroll
        for (int ks = 0; ks < K_STEPS; ++ks) {
            acc = __builtin_amdgcn_wmma_f32_16x16x32_f16(
                false, afrag[ks].v, false, bfrag[ks].v,
                (short)0, acc, false, false);
        }

        const int col = jt * 16 + rm;
        #pragma unroll
        for (int v = 0; v < 8; ++v) {
            const float c = acc[v];
            if (c > best[v]) { best[v] = c; bestj[v] = col; }  // first-max wins
        }

#if USE_TDM
        if (wid == 0 && s + 1 < JT_PER_CHUNK) {
            __builtin_amdgcn_s_wait_tensorcnt(0);
        }
        __syncthreads();   // also waits each wave's own ds loads (DScnt)
#else
        __syncthreads();
        if (s + 1 < JT_PER_CHUNK) {
            for (int c = threadIdx.x; c < 512; c += 256) {
                const int row = c >> 5, off = (c & 31) * 8;
                *(uint4*)(&sB[buf ^ 1][row * LDS_ROW_H + off]) =
                    *(const uint4*)(dstM +
                        (size_t)((jt + 1) * 16 + row) * D_DIM + off);
            }
            __syncthreads();
        }
#endif
    }

    // --- intra-wave argmax reduce (each wave's rows are disjoint) ---
    #pragma unroll
    for (int m = 8; m >= 1; m >>= 1) {
        #pragma unroll
        for (int v = 0; v < 8; ++v) {
            const float ov = __shfl_xor(best[v],  m, 32);
            const int   oj = __shfl_xor(bestj[v], m, 32);
            if (ov > best[v] || (ov == best[v] && oj < bestj[v])) {
                best[v] = ov; bestj[v] = oj;
            }
        }
    }
    if (rm == 0) {   // lanes 0 & 16 own rows hsel*8 + v
        #pragma unroll
        for (int v = 0; v < 8; ++v) {
            const int row = itile * 16 + hsel * 8 + v;
            const size_t o = (size_t)(t * JCHUNKS + jc) * N_PTS + row;
            pval[o] = best[v];
            pidx[o] = bestj[v];
        }
    }
}

// ---------------------------------------------------------------------------
// Kernel 3: reduce the JCHUNKS partials per row, gather matched points.
// ---------------------------------------------------------------------------
__global__ __launch_bounds__(256) void finalize_kernel(
    const float* __restrict__ pval, const int* __restrict__ pidx,
    const float* __restrict__ points,
    float* __restrict__ matched, float* __restrict__ conf)
{
    const int idx = blockIdx.x * 256 + threadIdx.x;
    if (idx >= (T_FRAMES - 1) * N_PTS) return;
    const int t = idx / N_PTS, row = idx % N_PTS;

    float bv = -3.0e38f;
    int   bj = 0x7fffffff;
    #pragma unroll
    for (int c = 0; c < JCHUNKS; ++c) {   // chunks cover ascending j ranges
        const size_t o = (size_t)(t * JCHUNKS + c) * N_PTS + row;
        const float v = pval[o];
        const int   j = pidx[o];
        if (v > bv || (v == bv && j < bj)) { bv = v; bj = j; }
    }
    conf[idx] = bv;
    const float2 pt =
        *(const float2*)(points + ((size_t)(t + 1) * N_PTS + bj) * 2);
    *(float2*)(matched + (size_t)idx * 2) = pt;
}

// ---------------------------------------------------------------------------
extern "C" void kernel_launch(void* const* d_in, const int* in_sizes, int n_in,
                              void* d_out, int out_size, void* d_ws, size_t ws_size,
                              hipStream_t stream)
{
    const float* desc = (const float*)d_in[0];   // [8][2048][256] f32
    const float* pts  = (const float*)d_in[1];   // [8][2048][2]   f32

    // workspace layout
    _Float16* nd = (_Float16*)d_ws;                              // 8 MB
    char* base   = (char*)d_ws + (size_t)T_FRAMES * N_PTS * D_DIM * 2;
    float* pval  = (float*)base;                                 // 224 KB
    int*   pidx  = (int*)(base +
                   (size_t)(T_FRAMES - 1) * JCHUNKS * N_PTS * sizeof(float));

    float* matched = (float*)d_out;                              // [7][2048][2]
    float* conf    = matched + (size_t)(T_FRAMES - 1) * N_PTS * 2;

    normalize_f16_kernel<<<(T_FRAMES * N_PTS) / NWAVES, 256, 0, stream>>>(desc, nd);
    match_wmma_kernel<<<(T_FRAMES - 1) * IB_PER_T * JCHUNKS, 256, 0, stream>>>(
        nd, pval, pidx);
    const int rows = (T_FRAMES - 1) * N_PTS;
    finalize_kernel<<<(rows + 255) / 256, 256, 0, stream>>>(
        pval, pidx, pts, matched, conf);
}
